// VectorQuantizer_249108103302
// MI455X (gfx1250) — matrix operand order repaired and checked by reading, not compiled, not run on hardware
//
#include <hip/hip_runtime.h>
#include <hip/hip_bf16.h>

typedef __attribute__((ext_vector_type(16))) __bf16 v16bf;
typedef __attribute__((ext_vector_type(8)))  float  v8f;

#define N_E   1024
#define CDIM  128
#define NQ    65536
#define HW    1024          // H*W
#define PLANE 131072        // C*H*W
#define QPB   128           // queries per block in main kernel
#define QS_STRIDE 130       // padded LDS row stride (bf16 elems) to spread banks
#define LOSS_OFF 8388608
#define IDX_OFF  8388609

__device__ __forceinline__ unsigned short f32_to_bf16_rne(float f) {
  unsigned int u = __float_as_uint(f);
  u += 0x7FFFu + ((u >> 16) & 1u);
  return (unsigned short)(u >> 16);
}

union BF16Frag { v16bf v; unsigned int u[8]; uint4 q[2]; };

// ---------------------------------------------------------------------------
// Kernel 1: embed = emb_w @ proj_w^T + b ; enorm = ||embed_row||^2 ;
//           embm2 = bf16(-2 * embed)   (fold -2 into the WMMA B operand)
// ---------------------------------------------------------------------------
__global__ void vq_embed_kernel(const float* __restrict__ emb_w,
                                const float* __restrict__ proj_w,
                                const float* __restrict__ proj_b,
                                float* __restrict__ embed,
                                float* __restrict__ enorm,
                                unsigned short* __restrict__ embm2) {
  __shared__ float s_row[CDIM];
  __shared__ float s_sq[CDIM];
  int n = blockIdx.x, c = threadIdx.x;
  s_row[c] = emb_w[n * CDIM + c];
  __syncthreads();
  float acc = proj_b[c];
  const float* pw = proj_w + c * CDIM;
#pragma unroll 8
  for (int k = 0; k < CDIM; ++k) acc = fmaf(s_row[k], pw[k], acc);
  embed[n * CDIM + c] = acc;
  embm2[n * CDIM + c] = f32_to_bf16_rne(-2.0f * acc);
  s_sq[c] = acc * acc;
  __syncthreads();
  for (int s = CDIM >> 1; s > 0; s >>= 1) {
    if (c < s) s_sq[c] += s_sq[c + s];
    __syncthreads();
  }
  if (c == 0) enorm[n] = s_sq[0];
}

// ---------------------------------------------------------------------------
// Kernel 2: distances via WMMA, argmin + online sum(exp(-d)), gather output.
// 256 threads = 8 waves; each wave owns a 16-row query tile; block owns 128
// consecutive queries (all within one batch image since 128 | 1024).
// Inner loop: TWO code tiles / iteration with independent accumulators so the
// interleaved WMMA stream has no back-to-back D->C hazard (no v_nop padding).
// ---------------------------------------------------------------------------
__global__ __launch_bounds__(256) void vq_main_kernel(
    const float* __restrict__ x,
    const float* __restrict__ embed,
    const float* __restrict__ enorm,
    const unsigned short* __restrict__ embm2,
    float* __restrict__ out,
    float* __restrict__ partials) {
  __shared__ unsigned short s_qs[QPB * QS_STRIDE];
  __shared__ float s_np[2 * QPB];
  __shared__ float s_qn[QPB];
  __shared__ int   s_idx[QPB];
  __shared__ float s_loss[QPB];

  const int t = threadIdx.x;
  const int qbase = blockIdx.x * QPB;
  const int b  = qbase / HW;
  const int r0 = qbase % HW;
  const float* xb = x + (size_t)b * PLANE + r0;

  // ---- stage 128 queries to LDS as bf16; accumulate ||q||^2 in f32 ----
  {
    int row = t & 127, seg = t >> 7;
    float nrm = 0.f;
#pragma unroll 4
    for (int c = seg * 64; c < seg * 64 + 64; ++c) {
      float v = xb[c * HW + row];           // coalesced across threads
      nrm = fmaf(v, v, nrm);
      s_qs[row * QS_STRIDE + c] = f32_to_bf16_rne(v);
    }
    s_np[seg * QPB + row] = nrm;
  }
  __syncthreads();
  if (t < QPB) s_qn[t] = s_np[t] + s_np[QPB + t];
  __syncthreads();

  const int wv   = t >> 5;
  const int lane = t & 31;
  const int col  = lane & 15;
  const int hlf  = lane >> 4;          // 0: lanes 0-15, 1: lanes 16-31
  const int m0   = wv * 16;
  const int mrow = m0 + col;

  // ---- A fragments, contiguous K: lanes 0-15 K 0..15, 16-31 K 16..31 (4 K-tiles of 32) ----
  BF16Frag afr[4];
#pragma unroll
  for (int kt = 0; kt < 4; ++kt) {
#pragma unroll
    for (int j = 0; j < 8; ++j) {
      int kl = 2 * j + (hlf ? 16 : 0);
      afr[kt].u[j] =
          *(const unsigned int*)&s_qs[mrow * QS_STRIDE + kt * 32 + kl];
    }
  }
  float qn[8];
#pragma unroll
  for (int j = 0; j < 8; ++j) qn[j] = s_qn[m0 + j + hlf * 8];

  float vmin[8], lse[8];
  int vidx[8];
#pragma unroll
  for (int j = 0; j < 8; ++j) { vmin[j] = 3.4e38f; lse[j] = 0.f; vidx[j] = 0; }

  const int kh = hlf * 16;             // B layout: lanes 0-15 K 0..15, 16-31 K 16..31
  for (int nb = 0; nb < N_E; nb += 32) {
    float en0 = enorm[nb + col];
    float en1 = enorm[nb + 16 + col];
    const unsigned short* brow0 = embm2 + (size_t)(nb + col) * CDIM;
    const unsigned short* brow1 = brow0 + 16 * CDIM;
    __builtin_prefetch(brow0 + 32 * CDIM, 0, 1);   // next pair of code tiles
    BF16Frag b0[4], b1[4];
#pragma unroll
    for (int kt = 0; kt < 4; ++kt) {
      const uint4* p0 = (const uint4*)(brow0 + kt * 32 + kh);  // 32B runs
      const uint4* p1 = (const uint4*)(brow1 + kt * 32 + kh);
      b0[kt].q[0] = p0[0];
      b0[kt].q[1] = p0[1];
      b1[kt].q[0] = p1[0];
      b1[kt].q[1] = p1[1];
    }
    v8f acc0 = {0.f, 0.f, 0.f, 0.f, 0.f, 0.f, 0.f, 0.f};
    v8f acc1 = {0.f, 0.f, 0.f, 0.f, 0.f, 0.f, 0.f, 0.f};
#pragma unroll
    for (int kt = 0; kt < 4; ++kt) {   // interleave: no back-to-back D->C reuse
      acc0 = __builtin_amdgcn_wmma_f32_16x16x32_bf16(
          false, afr[kt].v, false, b0[kt].v, (short)0, acc0, false, false);
      acc1 = __builtin_amdgcn_wmma_f32_16x16x32_bf16(
          false, afr[kt].v, false, b1[kt].v, (short)0, acc1, false, false);
    }
    // acc[j] = -2 q.e  for row M = j + hlf*8, col N = nb(+16) + col
#pragma unroll
    for (int j = 0; j < 8; ++j) {
      float d2a = qn[j] + en0 + acc0[j];
      float d2b = qn[j] + en1 + acc1[j];
      // args are O(100..500): raw v_sqrt_f32, skip libm denorm fixup
      float da = __builtin_amdgcn_sqrtf(fmaxf(d2a, 0.f));
      float db = __builtin_amdgcn_sqrtf(fmaxf(d2b, 0.f));
      lse[j] += __expf(-da) + __expf(-db);   // no shift: d_min is O(16)
      if (da < vmin[j]) { vmin[j] = da; vidx[j] = nb + col; }
      if (db < vmin[j]) { vmin[j] = db; vidx[j] = nb + 16 + col; }
    }
  }

  // ---- reduce across the 16 N-lanes of each half (min/argmin + sum) ----
#pragma unroll
  for (int j = 0; j < 8; ++j) {
#pragma unroll
    for (int off = 1; off < 16; off <<= 1) {
      float ov = __shfl_xor(vmin[j], off, 32);
      int   oi = __shfl_xor(vidx[j], off, 32);
      float os = __shfl_xor(lse[j],  off, 32);
      lse[j] += os;
      if (ov < vmin[j] || (ov == vmin[j] && oi < vidx[j])) {
        vmin[j] = ov; vidx[j] = oi;    // first-occurrence tie-break
      }
    }
  }
  if (col == 0) {
#pragma unroll
    for (int j = 0; j < 8; ++j) {
      int row = m0 + j + hlf * 8;
      s_idx[row]  = vidx[j];
      s_loss[row] = vmin[j] + __logf(lse[j]);  // d_min + log(sum exp(-d))
    }
  }
  __syncthreads();

  // ---- write x_q (coalesced over h,w) and indices ----
  {
    int row = t & 127, seg = t >> 7;
    int idx = s_idx[row];
    const float* er = embed + (size_t)idx * CDIM;
    float* ob = out + (size_t)b * PLANE + r0;
#pragma unroll 4
    for (int c = seg * 64; c < seg * 64 + 64; ++c)
      ob[c * HW + row] = er[c];
    if (seg == 0) out[IDX_OFF + qbase + row] = (float)idx;
  }

  // ---- deterministic per-block loss partial (no atomics) ----
  __syncthreads();
  for (int s = 64; s > 0; s >>= 1) {
    if (t < s) s_loss[t] += s_loss[t + s];
    __syncthreads();
  }
  if (t == 0) partials[blockIdx.x] = s_loss[0];
}

// ---------------------------------------------------------------------------
// Kernel 3: fixed-order reduction of 512 partials -> loss scalar
// ---------------------------------------------------------------------------
__global__ void vq_loss_reduce(const float* __restrict__ partials,
                               float* __restrict__ out) {
  __shared__ float s[256];
  int t = threadIdx.x;
  s[t] = partials[t] + partials[t + 256];
  __syncthreads();
  for (int k = 128; k > 0; k >>= 1) {
    if (t < k) s[t] += s[t + k];
    __syncthreads();
  }
  if (t == 0) out[LOSS_OFF] = s[0] * (1.0f / 65536.0f);
}

extern "C" void kernel_launch(void* const* d_in, const int* in_sizes, int n_in,
                              void* d_out, int out_size, void* d_ws, size_t ws_size,
                              hipStream_t stream) {
  const float* x      = (const float*)d_in[0];
  const float* emb_w  = (const float*)d_in[1];
  const float* proj_w = (const float*)d_in[2];
  const float* proj_b = (const float*)d_in[3];
  float* out = (float*)d_out;

  char* w = (char*)d_ws;
  float* embed = (float*)w;                                  // 1024*128 f32
  float* enorm = embed + N_E * CDIM;                         // 1024 f32
  unsigned short* embm2 = (unsigned short*)(enorm + N_E);    // 1024*128 bf16 (-2x)
  float* partials = (float*)(embm2 + N_E * CDIM);            // 512 f32

  vq_embed_kernel<<<N_E, CDIM, 0, stream>>>(emb_w, proj_w, proj_b,
                                            embed, enorm, embm2);
  vq_main_kernel<<<NQ / QPB, 256, 0, stream>>>(x, embed, enorm, embm2,
                                               out, partials);
  vq_loss_reduce<<<1, 256, 0, stream>>>(partials, out);
}
